// MultiHeadAttention_28870770163816
// MI455X (gfx1250) — matrix operand order repaired and checked by reading, not compiled, not run on hardware
//
#include <hip/hip_runtime.h>
#include <cmath>

// ---------------- fixed problem shape ----------------
constexpr int BATCH = 4;
constexpr int SEQ   = 2048;
constexpr int DM    = 1024;   // d_model
constexpr int NH    = 16;     // heads
constexpr int HD    = 64;     // head dim
constexpr int MROWS = BATCH * SEQ;   // 8192

typedef __attribute__((ext_vector_type(16))) __bf16 v16bf;
typedef __attribute__((ext_vector_type(8)))  __bf16 v8bf;
typedef __attribute__((ext_vector_type(8)))  float  v8f;

// ---------------- WMMA helper ----------------
__device__ __forceinline__ v8f wmma_bf16(v16bf a, v16bf b, v8f c) {
    return __builtin_amdgcn_wmma_f32_16x16x32_bf16(
        /*neg_a=*/false, a, /*neg_b=*/false, b,
        /*c_mod=*/(short)0, c, /*reuse_a=*/false, /*reuse_b=*/false);
}

// ---------------- DPP-fused butterfly reductions within each 16-lane half ----------------
// One VOP2+DPP16 instruction per step: dst = op(dpp(src), src)
__device__ __forceinline__ float red16_max(float x) {
    float y;
    asm("v_max_num_f32_dpp %0, %1, %1 quad_perm:[1,0,3,2] row_mask:0xf bank_mask:0xf bound_ctrl:1"
        : "=v"(y) : "v"(x));
    asm("v_max_num_f32_dpp %0, %1, %1 quad_perm:[2,3,0,1] row_mask:0xf bank_mask:0xf bound_ctrl:1"
        : "=v"(x) : "v"(y));
    asm("v_max_num_f32_dpp %0, %1, %1 row_half_mirror row_mask:0xf bank_mask:0xf bound_ctrl:1"
        : "=v"(y) : "v"(x));
    asm("v_max_num_f32_dpp %0, %1, %1 row_mirror row_mask:0xf bank_mask:0xf bound_ctrl:1"
        : "=v"(x) : "v"(y));
    return x;
}
__device__ __forceinline__ float red16_sum(float x) {
    float y;
    asm("v_add_f32_dpp %0, %1, %1 quad_perm:[1,0,3,2] row_mask:0xf bank_mask:0xf bound_ctrl:1"
        : "=v"(y) : "v"(x));
    asm("v_add_f32_dpp %0, %1, %1 quad_perm:[2,3,0,1] row_mask:0xf bank_mask:0xf bound_ctrl:1"
        : "=v"(x) : "v"(y));
    asm("v_add_f32_dpp %0, %1, %1 row_half_mirror row_mask:0xf bank_mask:0xf bound_ctrl:1"
        : "=v"(y) : "v"(x));
    asm("v_add_f32_dpp %0, %1, %1 row_mirror row_mask:0xf bank_mask:0xf bound_ctrl:1"
        : "=v"(x) : "v"(y));
    return x;
}

// A-fragment (16x32, 16-bit): lane<16 -> row, K {0..7}U{16..23}; lane>=16 -> K {8..15}U{24..31}
__device__ __forceinline__ v16bf load_a_bf16(const __bf16* base, int ld, int row, int k0) {
    int half = (threadIdx.x & 31) >> 4;
    const __bf16* p = base + (size_t)row * ld + k0 + half * 8;
    v8bf lo = *(const v8bf*)p;
    v8bf hi = *(const v8bf*)(p + 16);
    v16bf r;
#pragma unroll
    for (int i = 0; i < 8; ++i) { r[i] = lo[i]; r[i + 8] = hi[i]; }
    return r;
}

// B-fragment (32x16, 16-bit): lane<16 -> column=lane, K {0..7}U{16..23}; lane>=16 -> K {8..15}U{24..31}
__device__ __forceinline__ v16bf load_b_bf16(const __bf16* base, int ld, int row, int k0) {
    int half = (threadIdx.x & 31) >> 4;
    const __bf16* p = base + (size_t)row * ld + k0 + half * 8;
    v8bf lo = *(const v8bf*)p;
    v8bf hi = *(const v8bf*)(p + 16);
    v16bf r;
#pragma unroll
    for (int i = 0; i < 8; ++i) { r[i] = lo[i]; r[i + 8] = hi[i]; }
    return r;
}

// ---------------- f32 -> bf16 conversion (8 elements / thread, b128 in & out) ----------------
__global__ void cvt_f32_bf16_kernel(const float* __restrict__ in, __bf16* __restrict__ out, int n) {
    int i = (blockIdx.x * blockDim.x + threadIdx.x) * 8;
    if (i + 7 < n) {
        v8f v = *(const v8f*)(in + i);
        v8bf o;
#pragma unroll
        for (int j = 0; j < 8; ++j) o[j] = (__bf16)v[j];
        *(v8bf*)(out + i) = o;
    }
}

// ---------------- generic GEMM: C[M,N] = (A[M,K] * W[N,K]^T + bias) * oscale ----------------
// OMODE: 0 = bf16 [B,H,S,HD]  (Q / K layout)
//        1 = bf16 [B,H,HD,S]  (V transposed)
//        2 = f32  [M,N] row-major (final output)
template<int OMODE>
__global__ __launch_bounds__(256) void gemm_wmma_kernel(
    const __bf16* __restrict__ A, const __bf16* __restrict__ W,
    const float* __restrict__ bias, void* __restrict__ Out, float oscale)
{
    constexpr int K = DM, N = DM;
    int lane = threadIdx.x & 31;
    int wid  = threadIdx.x >> 5;       // 0..7
    int wm = wid >> 2;                 // 0..1
    int wn = wid & 3;                  // 0..3
    int m0 = (blockIdx.y * 2 + wm) * 64;
    int n0 = (blockIdx.x * 4 + wn) * 64;
    int r    = lane & 15;
    int half = lane >> 4;

    v8f acc[4][4];
#pragma unroll
    for (int tm = 0; tm < 4; ++tm)
#pragma unroll
        for (int tn = 0; tn < 4; ++tn)
#pragma unroll
            for (int i = 0; i < 8; ++i) acc[tm][tn][i] = 0.f;

    for (int k0 = 0; k0 < K; k0 += 32) {
        v16bf af[4], bfr[4];
#pragma unroll
        for (int t = 0; t < 4; ++t)
            af[t] = load_a_bf16(A, K, m0 + t * 16 + r, k0);
#pragma unroll
        for (int t = 0; t < 4; ++t)
            bfr[t] = load_b_bf16(W, K, n0 + t * 16 + r, k0);
#pragma unroll
        for (int tm = 0; tm < 4; ++tm)
#pragma unroll
            for (int tn = 0; tn < 4; ++tn)
                acc[tm][tn] = wmma_bf16(af[tm], bfr[tn], acc[tm][tn]);
    }

    // epilogue
#pragma unroll
    for (int tn = 0; tn < 4; ++tn) {
        int nn = n0 + tn * 16 + r;
        float bv = bias[nn];
        int h = nn >> 6, d = nn & 63;
#pragma unroll
        for (int tm = 0; tm < 4; ++tm) {
#pragma unroll
            for (int i = 0; i < 8; ++i) {
                int mm = m0 + tm * 16 + i + 8 * half;
                float val = (acc[tm][tn][i] + bv) * oscale;
                int b = mm >> 11;         // / SEQ
                int s = mm & (SEQ - 1);
                if (OMODE == 0) {
                    ((__bf16*)Out)[(((size_t)(b * NH + h) * SEQ + s) << 6) + d] = (__bf16)val;
                } else if (OMODE == 1) {
                    ((__bf16*)Out)[(((size_t)(b * NH + h) * HD + d) << 11) + s] = (__bf16)val;
                } else {
                    ((float*)Out)[(size_t)mm * N + nn] = val;
                }
            }
        }
    }
}

// ---------------- flash attention: one wave = 16 query rows of one (b,h) ----------------
// Q (pre-scaled by 1/sqrt(HD)), K: bf16 [B,H,S,64]; VT: bf16 [B,H,64,S]; AO: bf16 [B,S,DM]
__global__ __launch_bounds__(256) void attn_wmma_kernel(
    const __bf16* __restrict__ Q, const __bf16* __restrict__ Kt,
    const __bf16* __restrict__ VT, __bf16* __restrict__ AO)
{
    __shared__ __bf16 lds_p[8][16 * 64];

    int lane = threadIdx.x & 31;
    int wid  = threadIdx.x >> 5;
    int gw   = blockIdx.x * 8 + wid;          // 0 .. B*H*(S/16)-1
    int qt = gw & (SEQ / 16 - 1);             // 0..127
    int bh = gw >> 7;                         // 0..63
    int h = bh & (NH - 1);
    int b = bh >> 4;
    int r    = lane & 15;
    int half = lane >> 4;

    const size_t headOff = (size_t)(b * NH + h) * SEQ * HD;
    const __bf16* Qh = Q  + headOff;
    const __bf16* Kh = Kt + headOff;
    const __bf16* Vh = VT + headOff;          // [64][SEQ]

    v16bf aq[2];
    aq[0] = load_a_bf16(Qh, HD, qt * 16 + r, 0);
    aq[1] = load_a_bf16(Qh, HD, qt * 16 + r, 32);

    v8f m_run, l_run, acc[4];
#pragma unroll
    for (int i = 0; i < 8; ++i) { m_run[i] = -INFINITY; l_run[i] = 0.f; }
#pragma unroll
    for (int dt = 0; dt < 4; ++dt)
#pragma unroll
        for (int i = 0; i < 8; ++i) acc[dt][i] = 0.f;

    const float L2E = 1.4426950408889634f;

    for (int kt = 0; kt < SEQ; kt += 64) {
        // ---- scores S = Q K^T (16 x 64), Q pre-scaled ----
        v8f st[4];
#pragma unroll
        for (int t = 0; t < 4; ++t)
#pragma unroll
            for (int i = 0; i < 8; ++i) st[t][i] = 0.f;

#pragma unroll
        for (int t = 0; t < 4; ++t) {
            v16bf b0 = load_b_bf16(Kh, HD, kt + t * 16 + r, 0);
            v16bf b1 = load_b_bf16(Kh, HD, kt + t * 16 + r, 32);
            st[t] = wmma_bf16(aq[0], b0, st[t]);
            st[t] = wmma_bf16(aq[1], b1, st[t]);
        }

        // ---- row max across 64 keys (DPP-fused butterflies) ----
        v8f m_new, corr;
#pragma unroll
        for (int i = 0; i < 8; ++i) {
            float t = fmaxf(fmaxf(st[0][i], st[1][i]), fmaxf(st[2][i], st[3][i]));
            t = red16_max(t);
            m_new[i] = fmaxf(m_run[i], t);
            corr[i]  = exp2f((m_run[i] - m_new[i]) * L2E);
        }

        // ---- P = exp(S - m), row sums ----
        v8f rsum;
#pragma unroll
        for (int i = 0; i < 8; ++i) rsum[i] = 0.f;
#pragma unroll
        for (int t = 0; t < 4; ++t)
#pragma unroll
            for (int i = 0; i < 8; ++i) {
                float p = exp2f((st[t][i] - m_new[i]) * L2E);
                st[t][i] = p;
                rsum[i] += p;
            }
#pragma unroll
        for (int i = 0; i < 8; ++i) {
            rsum[i]  = red16_sum(rsum[i]);
            l_run[i] = l_run[i] * corr[i] + rsum[i];
            m_run[i] = m_new[i];
        }
#pragma unroll
        for (int dt = 0; dt < 4; ++dt)
#pragma unroll
            for (int i = 0; i < 8; ++i) acc[dt][i] *= corr[i];

        // ---- P (C/D layout) -> LDS row-major [16][64] -> A-frag layout ----
        __bf16* lp = lds_p[wid];
#pragma unroll
        for (int t = 0; t < 4; ++t)
#pragma unroll
            for (int i = 0; i < 8; ++i)
                lp[(i + 8 * half) * 64 + t * 16 + r] = (__bf16)st[t][i];

        asm volatile("s_wait_dscnt 0x0" ::: "memory");

#pragma unroll
        for (int ks = 0; ks < 2; ++ks) {
            v16bf ap = load_a_bf16(lp, 64, r, ks * 32);
#pragma unroll
            for (int dt = 0; dt < 4; ++dt) {
                v16bf bv = load_b_bf16(Vh, SEQ, dt * 16 + r, kt + ks * 32);
                acc[dt] = wmma_bf16(ap, bv, acc[dt]);
            }
        }
    }

    // ---- finalize: divide by row sums, write AO [B,S,DM] bf16 ----
    v8f inv;
#pragma unroll
    for (int i = 0; i < 8; ++i) inv[i] = 1.0f / l_run[i];

    size_t base = ((size_t)b * SEQ + qt * 16) * DM + h * HD;
#pragma unroll
    for (int dt = 0; dt < 4; ++dt)
#pragma unroll
        for (int i = 0; i < 8; ++i)
            AO[base + (size_t)(i + 8 * half) * DM + dt * 16 + r] = (__bf16)(acc[dt][i] * inv[i]);
}

// ---------------- host ----------------
extern "C" void kernel_launch(void* const* d_in, const int* in_sizes, int n_in,
                              void* d_out, int out_size, void* d_ws, size_t ws_size,
                              hipStream_t stream) {
    const float* query = (const float*)d_in[0];
    const float* key_  = (const float*)d_in[1];
    const float* value = (const float*)d_in[2];
    const float* Wq = (const float*)d_in[3];
    const float* bq = (const float*)d_in[4];
    const float* Wk = (const float*)d_in[5];
    const float* bk = (const float*)d_in[6];
    const float* Wv = (const float*)d_in[7];
    const float* bv = (const float*)d_in[8];
    const float* Wo = (const float*)d_in[9];
    const float* bo = (const float*)d_in[10];
    float* out = (float*)d_out;

    // workspace layout (bf16 elements)
    __bf16* ws  = (__bf16*)d_ws;
    const size_t WSZ = (size_t)DM * DM;               // 1M
    const size_t QSZ = (size_t)BATCH * NH * SEQ * HD; // 8M
    __bf16* wqb = ws;
    __bf16* wkb = wqb + WSZ;
    __bf16* wvb = wkb + WSZ;
    __bf16* wob = wvb + WSZ;
    __bf16* Qb  = wob + WSZ;
    __bf16* Kb  = Qb + QSZ;
    __bf16* VTb = Kb + QSZ;
    __bf16* AOb = VTb + QSZ;
    __bf16* Xb  = AOb + QSZ;   // shared activation staging buffer (reused 3x)

    const int nW = DM * DM;                 // 1M elems
    const int nX = MROWS * DM;              // 8M elems
    const int gW = nW / (256 * 8);          // 512 blocks
    const int gX = nX / (256 * 8);          // 4096 blocks

    // 1) weights f32 -> bf16
    cvt_f32_bf16_kernel<<<gW, 256, 0, stream>>>(Wq, wqb, nW);
    cvt_f32_bf16_kernel<<<gW, 256, 0, stream>>>(Wk, wkb, nW);
    cvt_f32_bf16_kernel<<<gW, 256, 0, stream>>>(Wv, wvb, nW);
    cvt_f32_bf16_kernel<<<gW, 256, 0, stream>>>(Wo, wob, nW);

    // 2) projections: stage each activation as bf16 (stream-ordered buffer reuse)
    dim3 gg(DM / 256, MROWS / 128);  // (4, 64); wave tile 64x64, WG = 128(M) x 256(N)
    const float qscale = 0.125f;     // 1/sqrt(HD), folded into Q projection
    cvt_f32_bf16_kernel<<<gX, 256, 0, stream>>>(query, Xb, nX);
    gemm_wmma_kernel<0><<<gg, 256, 0, stream>>>(Xb, wqb, bq, Qb, qscale);
    cvt_f32_bf16_kernel<<<gX, 256, 0, stream>>>(key_, Xb, nX);
    gemm_wmma_kernel<0><<<gg, 256, 0, stream>>>(Xb, wkb, bk, Kb, 1.0f);
    cvt_f32_bf16_kernel<<<gX, 256, 0, stream>>>(value, Xb, nX);
    gemm_wmma_kernel<1><<<gg, 256, 0, stream>>>(Xb, wvb, bv, VTb, 1.0f);

    // 3) attention: B*H*(S/16) = 8192 waves, 8 per block
    attn_wmma_kernel<<<(BATCH * NH * (SEQ / 16)) / 8, 256, 0, stream>>>(Qb, Kb, VTb, AOb);

    // 4) output projection: bf16 A, f32 out + bias
    gemm_wmma_kernel<2><<<gg, 256, 0, stream>>>(AOb, wob, bo, out, 1.0f);
}
